// _EmbeddingBlock_48455821033947
// MI455X (gfx1250) — compile-verified, matrix operand order checked by reading, not run on hardware
//
#include <hip/hip_runtime.h>
#include <stdint.h>

#define N_NODES  50000
#define NODE_DIM 128
#define N_EDGES  800000
#define EDGE_DIM 64
#define NRBF     16
#define H        64

typedef __bf16 bf16x16 __attribute__((ext_vector_type(16)));
typedef float  f32x8   __attribute__((ext_vector_type(8)));

union Frag { uint32_t u[8]; uint4 q[2]; bf16x16 v; };
union Acc  { f32x8 v; float f[8]; };

// round-to-nearest-even f32 -> bf16, packed pair {lo=a, hi=b}
__device__ __forceinline__ uint32_t pack_bf16(float a, float b) {
    uint32_t ua = __builtin_bit_cast(uint32_t, a);
    uint32_t ub = __builtin_bit_cast(uint32_t, b);
    ua += 0x7FFFu + ((ua >> 16) & 1u);
    ub += 0x7FFFu + ((ub >> 16) & 1u);
    return (ua >> 16) | (ub & 0xFFFF0000u);
}

// A fragment (16x32 bf16) from an LDS row of packed pairs: two ds_load_b128.
// ISA 16-bit A layout: lane holds row m=lane&15; VGPR v holds k-pair
// j = kc*16 + (lane>=16?4:0) + (v%4) + (v/4)*8.
__device__ __forceinline__ void load_afrag(Frag& f, const uint32_t* ldsRow, int kc, int lane) {
    int off = kc * 16 + ((lane >> 4) << 2);
    const uint4* p = (const uint4*)(ldsRow + off);
    f.q[0] = p[0];      // pairs jb .. jb+3
    f.q[1] = p[2];      // pairs jb+8 .. jb+11
}

// B fragment from fragment-major weight layout: lane's 8 dwords are contiguous
// -> two coalesced global_load_b128 (same bytes for every block: WGP$/L2 resident).
__device__ __forceinline__ void load_bfrag_g(Frag& f, const uint32_t* __restrict__ W,
                                             int fragIdx, int lane) {
    const uint4* p = (const uint4*)(W + ((fragIdx << 5) + lane) * 8);
    f.q[0] = p[0];
    f.q[1] = p[1];
}

__device__ __forceinline__ f32x8 wmma_bf16(const Frag& a, const Frag& b, f32x8 c) {
    return __builtin_amdgcn_wmma_f32_16x16x32_bf16(false, a.v, false, b.v, (short)0, c,
                                                   false, false);
}

// fragment-major index helpers: for frag (kc,c), lane L, vgpr v:
//   n = c*16 + (L&15);  j = kc*16 + ((L>>4)<<2) + (v&3) + ((v>>2)<<3)
__device__ __forceinline__ void frag_coords(int fragIdx, int t, int nPerKc, int& j, int& n) {
    int kc = fragIdx >> 2, c = fragIdx & 3;
    int lane = t >> 3, v = t & 7;
    n = c * 16 + (lane & 15);
    j = kc * 16 + ((lane >> 4) << 2) + (v & 3) + ((v >> 2) << 3);
    (void)nPerKc;
}

// ---------------------------------------------------------------------------
// Kernel 0: fuse weights through W_lin and emit fragment-major bf16 layouts.
//   We'[k][m] = sum_i W_edge[k][i] * W_lin[64+i][m]   (64x64)
//   Wr'[k][m] = sum_i W_rbf[k][i]  * W_lin[128+i][m]  (16x64, K-padded to 32)
//   b'[m]     = b_lin[m] + b_edge@W2 + b_rbf@W3
// ---------------------------------------------------------------------------
__global__ void prep_kernel(const float* __restrict__ Wn, const float* __restrict__ We,
                            const float* __restrict__ Wr, const float* __restrict__ Wl,
                            const float* __restrict__ be, const float* __restrict__ br,
                            const float* __restrict__ bl,
                            uint32_t* __restrict__ wnodeF, uint32_t* __restrict__ w1F,
                            uint32_t* __restrict__ weF, uint32_t* __restrict__ wrF,
                            float* __restrict__ bfused) {
    int tid = threadIdx.x;

    // W_node: 16 frags (kc 0..3 x c 0..3), plain pack
    for (int idx = tid; idx < 16 * 256; idx += 256) {
        int j, n; frag_coords(idx >> 8, idx & 255, 0, j, n);
        wnodeF[idx] = pack_bf16(Wn[(2 * j) * H + n], Wn[(2 * j + 1) * H + n]);
    }
    // W_lin rows 0..63 (P branch): 8 frags
    for (int idx = tid; idx < 8 * 256; idx += 256) {
        int j, n; frag_coords(idx >> 8, idx & 255, 0, j, n);
        w1F[idx] = pack_bf16(Wl[(2 * j) * H + n], Wl[(2 * j + 1) * H + n]);
    }
    // fused edge weights: 8 frags
    for (int idx = tid; idx < 8 * 256; idx += 256) {
        int j, n; frag_coords(idx >> 8, idx & 255, 0, j, n);
        float a = 0.f, b = 0.f;
        for (int i = 0; i < H; ++i) {
            float w2 = Wl[(64 + i) * H + n];
            a += We[(2 * j) * H + i] * w2;
            b += We[(2 * j + 1) * H + i] * w2;
        }
        weF[idx] = pack_bf16(a, b);
    }
    // fused rbf weights, K padded to 32: 4 frags (kc=0 only)
    for (int idx = tid; idx < 4 * 256; idx += 256) {
        int j, n; frag_coords(idx >> 8, idx & 255, 0, j, n);
        uint32_t pr = 0u;
        if (j < 8) {
            float a = 0.f, b = 0.f;
            for (int i = 0; i < H; ++i) {
                float w3 = Wl[(128 + i) * H + n];
                a += Wr[(2 * j) * H + i] * w3;
                b += Wr[(2 * j + 1) * H + i] * w3;
            }
            pr = pack_bf16(a, b);
        }
        wrF[idx] = pr;
    }
    // fused bias
    for (int n = tid; n < H; n += 256) {
        float s = bl[n];
        for (int i = 0; i < H; ++i) s += be[i] * Wl[(64 + i) * H + n];
        for (int i = 0; i < H; ++i) s += br[i] * Wl[(128 + i) * H + n];
        bfused[n] = s;
    }
}

// ---------------------------------------------------------------------------
// Kernel 1: h = x @ W_node + b_node  (50000 x 128 x 64), WMMA bf16, fp32 h out.
// Block = 128 threads (4 waves), 64 nodes per block, 16-row tile per wave.
// ---------------------------------------------------------------------------
__global__ void node_proj_kernel(const float* __restrict__ x,
                                 const uint32_t* __restrict__ wnodeF,
                                 const float* __restrict__ bnode,
                                 float* __restrict__ hbuf) {
    __shared__ __align__(16) uint32_t ldsX[64][68];   // 64 pairs/row, padded
    __shared__ float ldsO[64][68];
    int tid = threadIdx.x, lane = tid & 31, wave = tid >> 5;

    int base = blockIdx.x * 64;
    // Stage x tile as bf16 pairs: 2 rows per pass, coalesced 512B/row.
    for (int pass = 0; pass < 32; ++pass) {
        int row = pass * 2 + (tid >> 6);
        int j = tid & 63;
        int gn = base + row; if (gn >= N_NODES) gn = N_NODES - 1;
        float2 xv = ((const float2*)x)[(size_t)gn * 64 + j];
        ldsX[row][j] = pack_bf16(xv.x, xv.y);
    }
    __syncthreads();

    if (base + wave * 16 < N_NODES) {
        Acc acc[4];
#pragma unroll
        for (int c = 0; c < 4; ++c)
#pragma unroll
            for (int r = 0; r < 8; ++r) acc[c].f[r] = 0.f;
        int row = wave * 16 + (lane & 15);
#pragma unroll
        for (int kc = 0; kc < 4; ++kc) {
            Frag a; load_afrag(a, &ldsX[row][0], kc, lane);
#pragma unroll
            for (int c = 0; c < 4; ++c) {
                Frag b; load_bfrag_g(b, wnodeF, kc * 4 + c, lane);
                acc[c].v = wmma_bf16(a, b, acc[c].v);
            }
        }
        // D layout: lane n=c*16+(lane&15); VGPR r -> m = r + (lane>=16 ? 8 : 0)
        int mb = (lane >> 4) * 8;
#pragma unroll
        for (int c = 0; c < 4; ++c) {
            float bn = bnode[c * 16 + (lane & 15)];
#pragma unroll
            for (int r = 0; r < 8; ++r)
                ldsO[wave * 16 + mb + r][c * 16 + (lane & 15)] = acc[c].f[r] + bn;
        }
    }
    __syncthreads();
    int limit = (N_NODES - base) * H; if (limit > 64 * H) limit = 64 * H;
    for (int i = 0; i < 32; ++i) {
        int idx = i * 128 + tid;
        if (idx < limit) hbuf[(size_t)base * H + idx] = ldsO[idx >> 6][idx & 63];
    }
}

// ---------------------------------------------------------------------------
// Kernel 2: fused edge kernel.
// out = silu( (h[src]*h[dst]) @ W1 + edge_attr @ We' + rbf @ Wr' + b' )
// Block = 128 threads, 64 edges per block; 20 WMMAs per wave (5 K-chunks x 4 N).
// ---------------------------------------------------------------------------
__global__ void edge_fused_kernel(const float* __restrict__ hbuf,
                                  const float* __restrict__ eattr,
                                  const float* __restrict__ rbf,
                                  const long long* __restrict__ eidx,
                                  const uint32_t* __restrict__ w1F,
                                  const uint32_t* __restrict__ weF,
                                  const uint32_t* __restrict__ wrF,
                                  const float* __restrict__ bfused,
                                  float* __restrict__ out) {
    __shared__ __align__(16) uint32_t ldsP[64][36];   // xi*xj pairs (32 used)
    __shared__ __align__(16) uint32_t ldsE[64][36];   // edge_attr pairs
    __shared__ __align__(16) uint32_t ldsR[64][20];   // rbf pairs (8 real + 8 zero)
    __shared__ float ldsO[64][68];
    __shared__ int lsrc[64], ldst[64];
    int tid = threadIdx.x, lane = tid & 31, wave = tid >> 5;

    int base = blockIdx.x * 64;
    if (tid < 64) {
        lsrc[tid] = (int)eidx[base + tid];
        ldst[tid] = (int)eidx[N_EDGES + base + tid];
    }
    __syncthreads();

    // Stage P (gather xi,xj from L2-resident h table) and edge_attr.
    for (int pass = 0; pass < 16; ++pass) {
        int e = pass * 4 + (tid >> 5);
        int j = tid & 31;
        int s = lsrc[e], d = ldst[e];
        float2 a = ((const float2*)hbuf)[(size_t)s * 32 + j];
        float2 b = ((const float2*)hbuf)[(size_t)d * 32 + j];
        ldsP[e][j] = pack_bf16(a.x * b.x, a.y * b.y);
        float2 ea = ((const float2*)eattr)[(size_t)(base + e) * 32 + j];
        ldsE[e][j] = pack_bf16(ea.x, ea.y);
    }
    // Stage rbf (K=16 real, zero-pad to 32).
    for (int pass = 0; pass < 4; ++pass) {
        int e = pass * 16 + (tid >> 3);
        int j = tid & 7;
        float2 rv = ((const float2*)rbf)[(size_t)(base + e) * 8 + j];
        ldsR[e][j] = pack_bf16(rv.x, rv.y);
        ldsR[e][8 + j] = 0u;
    }
    __syncthreads();

    Acc acc[4];
#pragma unroll
    for (int c = 0; c < 4; ++c)
#pragma unroll
        for (int r = 0; r < 8; ++r) acc[c].f[r] = 0.f;

    int row = wave * 16 + (lane & 15);
#pragma unroll
    for (int kc = 0; kc < 2; ++kc) {            // P branch (K=64)
        Frag a; load_afrag(a, &ldsP[row][0], kc, lane);
#pragma unroll
        for (int c = 0; c < 4; ++c) {
            Frag b; load_bfrag_g(b, w1F, kc * 4 + c, lane);
            acc[c].v = wmma_bf16(a, b, acc[c].v);
        }
    }
#pragma unroll
    for (int kc = 0; kc < 2; ++kc) {            // fused edge branch (K=64)
        Frag a; load_afrag(a, &ldsE[row][0], kc, lane);
#pragma unroll
        for (int c = 0; c < 4; ++c) {
            Frag b; load_bfrag_g(b, weF, kc * 4 + c, lane);
            acc[c].v = wmma_bf16(a, b, acc[c].v);
        }
    }
    {                                           // fused rbf branch (K=32, padded)
        Frag a; load_afrag(a, &ldsR[row][0], 0, lane);
#pragma unroll
        for (int c = 0; c < 4; ++c) {
            Frag b; load_bfrag_g(b, wrF, c, lane);
            acc[c].v = wmma_bf16(a, b, acc[c].v);
        }
    }

    // bias + silu, transpose through LDS for coalesced stores
    int mb = (lane >> 4) * 8;
#pragma unroll
    for (int c = 0; c < 4; ++c) {
        float bn = bfused[c * 16 + (lane & 15)];
#pragma unroll
        for (int r = 0; r < 8; ++r) {
            float v = acc[c].f[r] + bn;
            v = v * (1.f / (1.f + __expf(-v)));
            ldsO[wave * 16 + mb + r][c * 16 + (lane & 15)] = v;
        }
    }
    __syncthreads();
    for (int i = 0; i < 32; ++i) {
        int idx = i * 128 + tid;
        out[(size_t)base * H + idx] = ldsO[idx >> 6][idx & 63];
    }
}

extern "C" void kernel_launch(void* const* d_in, const int* in_sizes, int n_in,
                              void* d_out, int out_size, void* d_ws, size_t ws_size,
                              hipStream_t stream) {
    (void)in_sizes; (void)n_in; (void)out_size; (void)ws_size;
    const float* x        = (const float*)d_in[0];
    const float* eattr    = (const float*)d_in[1];
    const float* rbf      = (const float*)d_in[2];
    const long long* eidx = (const long long*)d_in[3];
    const float* Wn = (const float*)d_in[4];
    const float* bn = (const float*)d_in[5];
    const float* We = (const float*)d_in[6];
    const float* be = (const float*)d_in[7];
    const float* Wr = (const float*)d_in[8];
    const float* br = (const float*)d_in[9];
    const float* Wl = (const float*)d_in[10];
    const float* bl = (const float*)d_in[11];
    float* out = (float*)d_out;

    // workspace layout (fragment-major weight blobs)
    float*    hbuf   = (float*)d_ws;                                // 50000*64 f32
    uint32_t* wnodeF = (uint32_t*)(hbuf + (size_t)N_NODES * H);     // 16*256 dw
    uint32_t* w1F    = wnodeF + 16 * 256;                           // 8*256 dw
    uint32_t* weF    = w1F + 8 * 256;                               // 8*256 dw
    uint32_t* wrF    = weF + 8 * 256;                               // 4*256 dw
    float*    bfused = (float*)(wrF + 4 * 256);                     // 64 f32

    hipLaunchKernelGGL(prep_kernel, dim3(1), dim3(256), 0, stream,
                       Wn, We, Wr, Wl, be, br, bl, wnodeF, w1F, weF, wrF, bfused);
    hipLaunchKernelGGL(node_proj_kernel, dim3((N_NODES + 63) / 64), dim3(128), 0, stream,
                       x, wnodeF, bn, hbuf);
    hipLaunchKernelGGL(edge_fused_kernel, dim3(N_EDGES / 64), dim3(128), 0, stream,
                       hbuf, eattr, rbf, eidx, w1F, weF, wrF, bfused, out);
}
